// LSTMModel_58952721104935
// MI455X (gfx1250) — matrix operand order repaired and checked by reading, not compiled, not run on hardware
//
#include <hip/hip_runtime.h>

typedef _Float16 v8h  __attribute__((ext_vector_type(8)));
typedef _Float16 v16h __attribute__((ext_vector_type(16)));
typedef float    v8f  __attribute__((ext_vector_type(8)));

#define SEQ_LEN  96
#define PRED_LEN 16
#define HIDDEN   64
#define FEAT     7
#define KDIM     96      // 64 (h) + 7 (x) + 25 (zero pad) -> 3 WMMA K-chunks of 32
#define NGATE    256     // 4*HIDDEN, PyTorch gate order i,f,g,o
#define ROWS     16      // batch rows per workgroup (one WMMA M-tile)

// ---------------------------------------------------------------------------
// Branchless fast tanh: prefer hardware v_tanh_f32 (gfx1250 TRANS op),
// else v_rcp_f32 + v_exp_f32 (both 1-ulp-class HW approximations).
// ---------------------------------------------------------------------------
__device__ __forceinline__ float fast_tanh(float x) {
#if __has_builtin(__builtin_amdgcn_tanhf)
    return __builtin_amdgcn_tanhf(x);
#else
    // tanh(x) = 2/(1+exp(-2x)) - 1 ; exp(-2x) = exp2(-2*log2e * x)
    return 2.f * __builtin_amdgcn_rcpf(
               1.f + __builtin_amdgcn_exp2f(-2.8853900817779268f * x)) - 1.f;
#endif
}

// ---------------------------------------------------------------------------
// Prep: pack [W_hh^T | W_ih^T | 0] as f16 [N=256][K=96] row-major, and b_ih+b_hh.
// Layout chosen so each lane's B-fragment read is 16 contiguous K-values.
// ---------------------------------------------------------------------------
__global__ void prep_weights(const float* __restrict__ W_ih,   // [256][7]
                             const float* __restrict__ W_hh,   // [256][64]
                             const float* __restrict__ b_ih,
                             const float* __restrict__ b_hh,
                             _Float16* __restrict__ WBT,       // [256][96] f16
                             float* __restrict__ bias)         // [256]
{
    int idx = blockIdx.x * blockDim.x + threadIdx.x;
    if (idx < NGATE * KDIM) {
        int n = idx / KDIM, k = idx % KDIM;
        float v = 0.f;
        if (k < HIDDEN)          v = W_hh[n * HIDDEN + k];
        else if (k < HIDDEN + 7) v = W_ih[n * FEAT + (k - HIDDEN)];
        WBT[idx] = (_Float16)v;
    }
    if (idx < NGATE) bias[idx] = b_ih[idx] + b_hh[idx];
}

// ---------------------------------------------------------------------------
// Persistent LSTM: one block owns 16 batch rows for the full 16x96 recurrence.
// Wave w owns gate w (i,f,g,o). Weights + bias register-resident; c in regs.
// ---------------------------------------------------------------------------
__global__ __launch_bounds__(128) void lstm_persistent(
    const float* __restrict__ x,        // [512][96][7]
    const _Float16* __restrict__ WBT,   // [256][96] f16
    const float* __restrict__ bias,     // [256]
    const float* __restrict__ fcW,      // [7][64]
    const float* __restrict__ fcB,      // [7]
    float* __restrict__ out)            // [512][16][7]
{
    __shared__ _Float16 hA[ROWS][KDIM];        // A operand: [h | x_t | 0]
    __shared__ _Float16 XW[112][ROWS][8];      // sliding x window (+16 fed-back preds)
    __shared__ float    gatesLDS[ROWS][NGATE]; // activated gates
    __shared__ float    hF32[ROWS][HIDDEN];    // f32 h for the FC head

    const int tid   = threadIdx.x;
    const int lane  = tid & 31;
    const int wv    = tid >> 5;         // wave id == gate id (0:i 1:f 2:g 3:o)
    const int hl    = lane >> 4;        // lane half (WMMA 16-bit operand layout)
    const int ll    = lane & 15;
    const int rbase = blockIdx.x * ROWS;

    // Unified branchless activation: act(v) = s0 * tanh(k*v) + s1
    //   sigmoid(v) = 0.5*tanh(0.5v)+0.5   (gates i,f,o)
    //   tanh(v)    = 1.0*tanh(1.0v)+0.0   (gate g)
    const float ak = (wv == 2) ? 1.0f : 0.5f;
    const float as = (wv == 2) ? 1.0f : 0.5f;
    const float ab = (wv == 2) ? 0.0f : 0.5f;

    // ---- one-time init -----------------------------------------------------
    for (int i = tid; i < ROWS * KDIM; i += 128)
        ((_Float16*)hA)[i] = (_Float16)0.f;          // h=0, pad cols stay 0 forever
    for (int i = tid; i < ROWS * HIDDEN; i += 128)
        ((float*)hF32)[i] = 0.f;
    for (int i = tid; i < SEQ_LEN * ROWS * FEAT; i += 128) {
        int t = i / (ROWS * FEAT);
        int r = (i / FEAT) % ROWS;
        int f = i % FEAT;
        XW[t][r][f] = (_Float16)x[(size_t)(rbase + r) * SEQ_LEN * FEAT + t * FEAT + f];
    }

    // prefetch weight pack into cache (gfx1250 global_prefetch path)
    for (int off = lane * 2048; off < NGATE * KDIM * 2; off += 32 * 2048)
        __builtin_prefetch((const char*)WBT + off, 0, 1);

    // ---- B fragments + bias: register resident for the whole kernel -------
    // B (f16, 32x16 K x N): lanes 0-15 hold K 0..15 of col ll, lanes 16-31 hold K 16..31.
    v16h B[4][3];
    float bv[4];
    #pragma unroll
    for (int n = 0; n < 4; ++n) {
        const int N = wv * 64 + n * 16 + ll;
        bv[n] = bias[N];
        #pragma unroll
        for (int kk = 0; kk < 3; ++kk)
            B[n][kk] = *(const v16h*)(WBT + (size_t)N * KDIM + kk * 32 + hl * 16);
    }

    float creg[8];                       // cell state: thread owns rows tid/8, cols (tid%8)*8..+7
    #pragma unroll
    for (int j = 0; j < 8; ++j) creg[j] = 0.f;

    const int ur = tid >> 3;             // update-phase row
    const int uc = (tid & 7) * 8;        // update-phase col base (contiguous 8)

    __syncthreads();
    // splice x_0 of window 0 into A cols 64..70
    if (tid < ROWS * FEAT)
        hA[tid / FEAT][HIDDEN + tid % FEAT] = XW[0][tid / FEAT][tid % FEAT];

    for (int p = 0; p < PRED_LEN; ++p) {
        for (int t = 0; t < SEQ_LEN; ++t) {
            __syncthreads();   // h + x_t splice visible

            // A fragments (f16 16x32): lane row = ll; two 16B chunks per K-slice
            v16h A[3];
            #pragma unroll
            for (int kk = 0; kk < 3; ++kk) {
                const _Float16* src = &hA[ll][kk * 32 + hl * 16];
                v8h lo = *(const v8h*)(src);
                v8h hi = *(const v8h*)(src + 8);
                v16h a;
                #pragma unroll
                for (int e = 0; e < 8; ++e) { a[e] = lo[e]; a[8 + e] = hi[e]; }
                A[kk] = a;
            }

            // gates = [h|x|0] @ W^T + b : 4 N-tiles x 3 K-chunks = 12 WMMA per wave
            #pragma unroll
            for (int n = 0; n < 4; ++n) {
                v8f acc;
                #pragma unroll
                for (int e = 0; e < 8; ++e) acc[e] = bv[n];
                #pragma unroll
                for (int kk = 0; kk < 3; ++kk)
                    acc = __builtin_amdgcn_wmma_f32_16x16x32_f16(
                              false, A[kk], false, B[n][kk],
                              (short)0, acc, false, false);
                // branchless activation in accumulator layout, spill to LDS
                #pragma unroll
                for (int e = 0; e < 8; ++e) {
                    float a = as * fast_tanh(ak * acc[e]) + ab;
                    gatesLDS[e + 8 * hl][wv * 64 + n * 16 + ll] = a;
                }
            }
            __syncthreads();   // gates visible

            // cell/hidden update: contiguous 8 cols per thread -> b128-mergeable
            #pragma unroll
            for (int j = 0; j < 8; ++j) {
                int cc = uc + j;
                float ig = gatesLDS[ur][cc];
                float fg = gatesLDS[ur][64 + cc];
                float gg = gatesLDS[ur][128 + cc];
                float og = gatesLDS[ur][192 + cc];
                float cn = fg * creg[j] + ig * gg;
                float hn = og * fast_tanh(cn);
                creg[j] = cn;
                hF32[ur][cc] = hn;
                hA[ur][cc] = (_Float16)hn;
            }
            // splice next x_t (disjoint hA cols 64..70 -> same barrier region)
            if (t < SEQ_LEN - 1 && tid < ROWS * FEAT)
                hA[tid / FEAT][HIDDEN + tid % FEAT] = XW[p + t + 1][tid / FEAT][tid % FEAT];
        }
        __syncthreads();   // hF32 visible

        // pred = h @ fcW^T + fcB (16x7): write out, feed back, splice next window's x_0
        if (tid < ROWS * FEAT) {
            int r = tid / FEAT, cf = tid % FEAT;
            float s = fcB[cf];
            #pragma unroll 8
            for (int k = 0; k < HIDDEN; ++k) s += hF32[r][k] * fcW[cf * HIDDEN + k];
            out[(size_t)(rbase + r) * PRED_LEN * FEAT + p * FEAT + cf] = s;
            XW[SEQ_LEN + p][r][cf] = (_Float16)s;
            if (p < PRED_LEN - 1)
                hA[r][HIDDEN + cf] = XW[p + 1][r][cf];
        }
    }
}

// ---------------------------------------------------------------------------
extern "C" void kernel_launch(void* const* d_in, const int* in_sizes, int n_in,
                              void* d_out, int out_size, void* d_ws, size_t ws_size,
                              hipStream_t stream) {
    const float* x    = (const float*)d_in[0];
    const float* W_ih = (const float*)d_in[1];
    const float* W_hh = (const float*)d_in[2];
    const float* b_ih = (const float*)d_in[3];
    const float* b_hh = (const float*)d_in[4];
    const float* fc_W = (const float*)d_in[5];
    const float* fc_b = (const float*)d_in[6];

    _Float16* WBT  = (_Float16*)d_ws;                          // 256*96*2 = 49152 B
    float*    bias = (float*)((char*)d_ws + NGATE * KDIM * 2); // 1024 B

    prep_weights<<<(NGATE * KDIM + 255) / 256, 256, 0, stream>>>(
        W_ih, W_hh, b_ih, b_hh, WBT, bias);

    lstm_persistent<<<512 / ROWS, 128, 0, stream>>>(
        x, WBT, bias, fc_W, fc_b, (float*)d_out);
}
